// GroupedExpertsHF_22892175688405
// MI455X (gfx1250) — matrix operand order repaired and checked by reading, not compiled, not run on hardware
//
#include <hip/hip_runtime.h>
#include <cstdint>
#include <cstddef>

// ---------------------------------------------------------------------------
// Grouped-expert SwiGLU for MI455X (gfx1250, wave32, WMMA bf16).
//   h   = silu(x @ G_e^T) * (x @ U_e^T)      [16384 x 1024]  (bf16, in d_ws)
//   out = h @ D_e^T                          [16384 x 2048]  (fp32, d_out)
// Double-buffered LDS pipeline: global loads for tile k+1 are issued into
// VGPR staging at the top of iteration k, WMMAs run on LDS buffer `cur`,
// staged registers are converted+stored into buffer cur^1, ONE barrier/iter.
// All expert token counts are multiples of 128 -> a 128-row tile never
// straddles an expert boundary.
// ---------------------------------------------------------------------------

typedef __attribute__((ext_vector_type(16))) __bf16 v16bf;
typedef __attribute__((ext_vector_type(8)))  float  v8f;

#define DIM_    2048
#define HID_    1024
#define NEXP_   8
#define TOTAL_  16384

#define BM    128     // block tile M (tokens)
#define BNA   64      // kernel-1 block tile N (hidden features)
#define BNB   128     // kernel-2 block tile N (output features)
#define KT    32      // K step per WMMA (bf16 K=32)
#define LDT   40      // padded LDS row stride in elements (80B = 20 banks)

__device__ __forceinline__ int find_expert(const int* __restrict__ counts, int m0) {
  int e = 0;
  int acc = counts[0];
  while (e < NEXP_ - 1 && m0 >= acc) { ++e; acc += counts[e]; }
  return e;
}

__device__ __forceinline__ void cvt_store4(__bf16* dst, float4 v) {
  dst[0] = (__bf16)v.x; dst[1] = (__bf16)v.y;
  dst[2] = (__bf16)v.z; dst[3] = (__bf16)v.w;
}

// A-matrix fragment (16-bit 16x32, contiguous K order, not the ISA table): lane half hs selects
// K = hs*16..hs*16+15 -> two 16B LDS reads at +0 / +16 bytes.
__device__ __forceinline__ v16bf load_frag_a(const __bf16* base) {
  v16bf f;
  ((uint4*)&f)[0] = *(const uint4*)(base);
  ((uint4*)&f)[1] = *(const uint4*)(base + 8);
  return f;
}

// B-matrix fragment (dense KxN layout, contiguous K order as in A, not the ISA table): lane half selects 16
// consecutive K values -> two 16B LDS reads at +0 / +16 bytes.
__device__ __forceinline__ v16bf load_frag_b(const __bf16* base) {
  v16bf f;
  ((uint4*)&f)[0] = *(const uint4*)(base);
  ((uint4*)&f)[1] = *(const uint4*)(base + 8);
  return f;
}

// ---------------------------------------------------------------------------
// Kernel 1: gate/up GEMMs + fused SwiGLU. Writes bf16 h into workspace.
// Grid (HID_/BNA, TOTAL_/BM), 256 threads = 8 waves (4 M x 2 N),
// wave tile 32x32 with BOTH gate and up accumulators (8 WMMA / K-step).
// ---------------------------------------------------------------------------
extern "C" __global__ void __launch_bounds__(256)
moe_gateup_silu(const float* __restrict__ x,
                const float* __restrict__ gate,
                const float* __restrict__ up,
                const int*   __restrict__ counts,
                unsigned short* __restrict__ hbuf)
{
  __shared__ __bf16 As[2][BM  * LDT];
  __shared__ __bf16 Bg[2][BNA * LDT];
  __shared__ __bf16 Bu[2][BNA * LDT];

  const int tid = threadIdx.x;
  const int m0  = blockIdx.y * BM;
  const int n0  = blockIdx.x * BNA;

  const int e = find_expert(counts, m0);
  const float* __restrict__ G = gate + (size_t)e * HID_ * DIM_;
  const float* __restrict__ U = up   + (size_t)e * HID_ * DIM_;

  const int lane = tid & 31;
  const int wave = tid >> 5;
  const int wm   = (wave & 3) * 32;
  const int wn   = (wave >> 2) * 32;
  const int lr   = lane & 15;
  const int hs   = lane >> 4;

  const int lrow = tid >> 3;          // loader: 32 rows / pass
  const int lcol = (tid & 7) * 4;     // loader: 4 floats / thread

  float4 sx[4], sg[2], su[2];         // register staging for next K tile

  auto load_tiles = [&](int k0) {
#pragma unroll
    for (int p = 0; p < 4; ++p)
      sx[p] = *(const float4*)(x + (size_t)(m0 + lrow + 32 * p) * DIM_ + (k0 + lcol));
#pragma unroll
    for (int p = 0; p < 2; ++p) {
      sg[p] = *(const float4*)(G + (size_t)(n0 + lrow + 32 * p) * DIM_ + (k0 + lcol));
      su[p] = *(const float4*)(U + (size_t)(n0 + lrow + 32 * p) * DIM_ + (k0 + lcol));
    }
  };
  auto store_tiles = [&](int b) {
#pragma unroll
    for (int p = 0; p < 4; ++p)
      cvt_store4(&As[b][(lrow + 32 * p) * LDT + lcol], sx[p]);
#pragma unroll
    for (int p = 0; p < 2; ++p) {
      cvt_store4(&Bg[b][(lrow + 32 * p) * LDT + lcol], sg[p]);
      cvt_store4(&Bu[b][(lrow + 32 * p) * LDT + lcol], su[p]);
    }
  };

  v8f ag[2][2] = {};
  v8f au[2][2] = {};

  // prologue: tile 0 -> buf 0
  load_tiles(0);
  store_tiles(0);
  __syncthreads();

  int cur = 0;
  for (int k0 = 0; k0 < DIM_; k0 += KT) {
    const bool hasnext = (k0 + KT) < DIM_;
    if (hasnext) load_tiles(k0 + KT);   // in flight during WMMA phase

    v16bf af[2], fg[2], fu[2];
#pragma unroll
    for (int mt = 0; mt < 2; ++mt)
      af[mt] = load_frag_a(&As[cur][(wm + mt * 16 + lr) * LDT + hs * 16]);
#pragma unroll
    for (int nt = 0; nt < 2; ++nt) {
      fg[nt] = load_frag_b(&Bg[cur][(wn + nt * 16 + lr) * LDT + hs * 16]);
      fu[nt] = load_frag_b(&Bu[cur][(wn + nt * 16 + lr) * LDT + hs * 16]);
    }
#pragma unroll
    for (int mt = 0; mt < 2; ++mt)
#pragma unroll
      for (int nt = 0; nt < 2; ++nt) {
        ag[mt][nt] = __builtin_amdgcn_wmma_f32_16x16x32_bf16(
            false, af[mt], false, fg[nt], (short)0, ag[mt][nt], false, false);
        au[mt][nt] = __builtin_amdgcn_wmma_f32_16x16x32_bf16(
            false, af[mt], false, fu[nt], (short)0, au[mt][nt], false, false);
      }

    if (hasnext) {
      store_tiles(cur ^ 1);
      __syncthreads();                  // single barrier per iteration
      cur ^= 1;
    }
  }

  // fused SwiGLU epilogue, bf16 store
  __bf16* __restrict__ hb = (__bf16*)hbuf;
#pragma unroll
  for (int mt = 0; mt < 2; ++mt)
#pragma unroll
    for (int nt = 0; nt < 2; ++nt) {
      const int hid = n0 + wn + nt * 16 + lr;
#pragma unroll
      for (int i = 0; i < 8; ++i) {
        const int tok = m0 + wm + mt * 16 + hs * 8 + i;  // C/D: M = i + 8*hs
        float g = ag[mt][nt][i];
        float u = au[mt][nt][i];
        float s = g / (1.0f + __expf(-g));               // silu(g)
        hb[(size_t)tok * HID_ + hid] = (__bf16)(s * u);
      }
    }
}

// ---------------------------------------------------------------------------
// Kernel 2: down projection, out = h @ D_e^T (fp32).
// Grid (DIM_/BNB, TOTAL_/BM), 8 waves (4 M x 2 N), wave tile 32x64
// (2 M-tiles x 4 N-tiles -> 8 WMMA per K-step, 12 ds_load_b128).
// ---------------------------------------------------------------------------
extern "C" __global__ void __launch_bounds__(256)
moe_down(const unsigned short* __restrict__ hbuf,
         const float* __restrict__ down,
         const int*   __restrict__ counts,
         float* __restrict__ out)
{
  __shared__ __bf16 Ah[2][BM  * LDT];
  __shared__ __bf16 Bd[2][BNB * LDT];

  const int tid = threadIdx.x;
  const int m0  = blockIdx.y * BM;
  const int n0  = blockIdx.x * BNB;

  const int e = find_expert(counts, m0);
  const float*  __restrict__ D  = down + (size_t)e * DIM_ * HID_;
  const __bf16* __restrict__ hb = (const __bf16*)hbuf;

  const int lane = tid & 31;
  const int wave = tid >> 5;
  const int wm   = (wave & 3) * 32;
  const int wn   = (wave >> 2) * 64;
  const int lr   = lane & 15;
  const int hs   = lane >> 4;

  const int arow = tid >> 2;          // h loader: 64 rows / pass, 8 bf16 / thread
  const int acol = (tid & 3) * 8;
  const int brow = tid >> 3;          // weight loader: 32 rows / pass
  const int bcol = (tid & 7) * 4;

  uint4  sa[2];                       // staged h (raw bf16)
  float4 sb[4];                       // staged weights (fp32)

  auto load_tiles = [&](int k0) {
#pragma unroll
    for (int p = 0; p < 2; ++p)
      sa[p] = *(const uint4*)(hb + (size_t)(m0 + arow + 64 * p) * HID_ + (k0 + acol));
#pragma unroll
    for (int p = 0; p < 4; ++p)
      sb[p] = *(const float4*)(D + (size_t)(n0 + brow + 32 * p) * HID_ + (k0 + bcol));
  };
  auto store_tiles = [&](int b) {
#pragma unroll
    for (int p = 0; p < 2; ++p)
      *(uint4*)&Ah[b][(arow + 64 * p) * LDT + acol] = sa[p];
#pragma unroll
    for (int p = 0; p < 4; ++p)
      cvt_store4(&Bd[b][(brow + 32 * p) * LDT + bcol], sb[p]);
  };

  v8f acc[2][4] = {};

  load_tiles(0);
  store_tiles(0);
  __syncthreads();

  int cur = 0;
  for (int k0 = 0; k0 < HID_; k0 += KT) {
    const bool hasnext = (k0 + KT) < HID_;
    if (hasnext) load_tiles(k0 + KT);

    v16bf af[2], bf[4];
#pragma unroll
    for (int mt = 0; mt < 2; ++mt)
      af[mt] = load_frag_a(&Ah[cur][(wm + mt * 16 + lr) * LDT + hs * 16]);
#pragma unroll
    for (int nt = 0; nt < 4; ++nt)
      bf[nt] = load_frag_b(&Bd[cur][(wn + nt * 16 + lr) * LDT + hs * 16]);
#pragma unroll
    for (int mt = 0; mt < 2; ++mt)
#pragma unroll
      for (int nt = 0; nt < 4; ++nt)
        acc[mt][nt] = __builtin_amdgcn_wmma_f32_16x16x32_bf16(
            false, af[mt], false, bf[nt], (short)0, acc[mt][nt], false, false);

    if (hasnext) {
      store_tiles(cur ^ 1);
      __syncthreads();
      cur ^= 1;
    }
  }

#pragma unroll
  for (int mt = 0; mt < 2; ++mt)
#pragma unroll
    for (int nt = 0; nt < 4; ++nt) {
      const int col = n0 + wn + nt * 16 + lr;
#pragma unroll
      for (int i = 0; i < 8; ++i) {
        const int tok = m0 + wm + mt * 16 + hs * 8 + i;
        out[(size_t)tok * DIM_ + col] = acc[mt][nt][i];
      }
    }
}

// ---------------------------------------------------------------------------
// Host launcher. d_in order: x, gate_proj, up_proj, down_proj, counts.
// d_ws holds the bf16 intermediate h (16384*1024*2 = 32 MB).
// ---------------------------------------------------------------------------
extern "C" void kernel_launch(void* const* d_in, const int* in_sizes, int n_in,
                              void* d_out, int out_size, void* d_ws, size_t ws_size,
                              hipStream_t stream) {
  const float* x      = (const float*)d_in[0];
  const float* gate   = (const float*)d_in[1];
  const float* up     = (const float*)d_in[2];
  const float* down   = (const float*)d_in[3];
  const int*   counts = (const int*)d_in[4];

  unsigned short* hbuf = (unsigned short*)d_ws;  // bf16 h, fully written by K1
  float* outp = (float*)d_out;

  dim3 blk(256);
  dim3 gridA(HID_ / BNA, TOTAL_ / BM);  // (16, 128)
  dim3 gridB(DIM_ / BNB, TOTAL_ / BM);  // (16, 128)

  moe_gateup_silu<<<gridA, blk, 0, stream>>>(x, gate, up, counts, hbuf);
  moe_down<<<gridB, blk, 0, stream>>>(hbuf, down, counts, outp);
}
